// SUNConv_78572131713436
// MI455X (gfx1250) — matrix operand order repaired and checked by reading, not compiled, not run on hardware
//
#include <hip/hip_runtime.h>
#include <hip/hip_bf16.h>

// ---------------------------------------------------------------------------
// SUNConv fused pipeline for MI455X (gfx1250, wave32, bf16 WMMA f32-accum)
// B=8, N=128, D=OUT=64
// ---------------------------------------------------------------------------

typedef __attribute__((ext_vector_type(16))) __bf16 v16bf;
typedef __attribute__((ext_vector_type(8)))  __bf16 v8bf;
typedef __attribute__((ext_vector_type(8)))  float  v8f;

#define KB_B   8
#define KB_N   128
#define KB_D   64
#define ROWS   (KB_B * KB_N * KB_N)   // 131072 flattened (b,i,k) rows

// ------------------------- workspace layout (bytes) -------------------------
#define OFF_XBF     (size_t)0           // [ROWS][64] bf16          16,777,216
#define OFF_X0BF    (size_t)16777216    // [ROWS][64] bf16          16,777,216
#define OFF_YT      (size_t)33554432    // per (b,i): [64][128] bf16 16,777,216
#define OFF_ABFT    (size_t)50331648    // [b][j][k] bf16               262,144
#define OFF_WC0     (size_t)50593792    // [448][64] f32                114,688
#define OFF_WC1     (size_t)50708480    // [448][64] f32                114,688
#define OFF_W1T     (size_t)50823168    // [64][64] bf16 (transposed)     8,192
#define OFF_W4T     (size_t)50831360    // [64][64] bf16 (transposed)     8,192
#define OFF_LIN0T   (size_t)50839552    // [64][64] bf16 (transposed)     8,192
#define OFF_XDIAG   (size_t)50847744    // [b][n][d] f32                262,144
#define OFF_XMI     (size_t)51109888    // mean over i  [b][j][d]       262,144
#define OFF_XMJ     (size_t)51372032    // mean over j  [b][i][d]       262,144
#define OFF_X0MEAN  (size_t)51634176    // mean_i X0    [b][k][d]       262,144
#define OFF_X7M     (size_t)51896320    // A^T @ X0mean [b][j][d]       262,144
#define OFF_CI0     (size_t)52158464    // [b][i][n] f32 (+bias)        262,144
#define OFF_CJ0     (size_t)52420608    // [b][j][n] f32                262,144
#define OFF_X4DIAG  (size_t)52682752    // [b][i][d] f32                262,144

// ---------------------------------------------------------------------------
// WMMA tile loaders.
// A-operand (16x32, row-major source, row stride ld elems):
//   lane<16: M=lane,    holds K = {0..7} and {16..23}
//   lane>=16:M=lane-16, holds K = {8..15} and {24..31}
// B-operand (32x16) from transposed source Bt[n][k], row stride ld elems:
//   lane<16: N=lane,    K = {0..7} and {16..23}
//   lane>=16:N=lane-16, K = {8..15} and {24..31}
// ---------------------------------------------------------------------------
static __device__ __forceinline__ v16bf load_a32(const __bf16* base, int ld, int lane) {
    int half = lane >> 4;
    int m    = lane & 15;
    const __bf16* p = base + (size_t)m * ld + half * 8;
    v8bf lo = *(const v8bf*)p;         // K = K0 + half*8 + 0..7
    v8bf hi = *(const v8bf*)(p + 16);  // K = K0 + 16 + half*8 + 0..7
    return __builtin_shufflevector(lo, hi, 0,1,2,3,4,5,6,7,8,9,10,11,12,13,14,15);
}

static __device__ __forceinline__ v16bf load_bt32(const __bf16* bt, int ld, int lane) {
    int half = lane >> 4;
    int nn   = lane & 15;
    const __bf16* p = bt + (size_t)nn * ld + half * 8;
    v8bf lo = *(const v8bf*)p;
    v8bf hi = *(const v8bf*)(p + 16);
    return __builtin_shufflevector(lo, hi, 0,1,2,3,4,5,6,7,8,9,10,11,12,13,14,15);
}

static __device__ __forceinline__ v8f wmma_bf16(v16bf a, v16bf b, v8f c) {
    return __builtin_amdgcn_wmma_f32_16x16x32_bf16(false, a, false, b, (short)0, c,
                                                   false, false);
}

// ---------------------------------------------------------------------------
// Precompute / conversion kernels (tiny, fp32 scalar)
// ---------------------------------------------------------------------------
__global__ void k_prep_weights(const float* __restrict__ hw,     // [2][448][64]
                               const float* __restrict__ mlp1w,  // [64][64]
                               const float* __restrict__ lin0w,  // [64][64]
                               float* __restrict__ Wc0, float* __restrict__ Wc1,
                               __bf16* __restrict__ W1t, __bf16* __restrict__ W4t,
                               __bf16* __restrict__ lin0t) {
    int tid = threadIdx.x;
    for (int idx = tid; idx < 448 * 64; idx += 256) {
        int r = idx >> 6, n = idx & 63;
        float s0 = 0.f, s1 = 0.f;
        for (int m = 0; m < 64; ++m) {
            float w = mlp1w[m * 64 + n];
            s0 += hw[r * 64 + m] * w;
            s1 += hw[448 * 64 + r * 64 + m] * w;
        }
        Wc0[idx] = s0;
        Wc1[idx] = s1;
    }
    __syncthreads();
    for (int idx = tid; idx < 64 * 64; idx += 256) {
        int nn = idx >> 6, k = idx & 63;
        W1t[idx]   = (__bf16)Wc0[(0 * 64 + k) * 64 + nn];   // block 0 (X1 term)
        W4t[idx]   = (__bf16)Wc0[(3 * 64 + k) * 64 + nn];   // block 3 (X4 term)
        lin0t[idx] = (__bf16)lin0w[k * 64 + nn];
    }
}

__global__ void k_cvt_x(const float* __restrict__ X, __bf16* __restrict__ Xbf, int total) {
    int i = blockIdx.x * 256 + threadIdx.x;
    if (i < total) Xbf[i] = (__bf16)X[i];
}

__global__ void k_cvt_a(const float* __restrict__ A, __bf16* __restrict__ AbfT) {
    int idx = blockIdx.x * 256 + threadIdx.x;      // (b*128 + j)*128 + k
    if (idx >= KB_B * KB_N * KB_N) return;
    int b = idx >> 14, rem = idx & 16383, j = rem >> 7, k = rem & 127;
    AbfT[idx] = (__bf16)A[(size_t)(b * KB_N + k) * KB_N + j];
}

__global__ void k_stats1(const float* __restrict__ X, float* __restrict__ Xdiag,
                         float* __restrict__ Xmi, float* __restrict__ Xmj) {
    int t = blockIdx.x * 256 + threadIdx.x;        // (b*128 + node)*64 + d
    if (t >= KB_B * KB_N * KB_D) return;
    int b = t >> 13, rem = t & 8191, node = rem >> 6, d = rem & 63;
    size_t bb = (size_t)b * KB_N * KB_N * KB_D;
    Xdiag[t] = X[bb + ((size_t)node * KB_N + node) * KB_D + d];
    float s = 0.f;
    for (int i = 0; i < KB_N; ++i) s += X[bb + ((size_t)i * KB_N + node) * KB_D + d];
    Xmi[t] = s * (1.f / KB_N);
    s = 0.f;
    for (int j = 0; j < KB_N; ++j) s += X[bb + ((size_t)node * KB_N + j) * KB_D + d];
    Xmj[t] = s * (1.f / KB_N);
}

__global__ void k_x0mean(const __bf16* __restrict__ X0bf, float* __restrict__ X0mean) {
    int t = blockIdx.x * 256 + threadIdx.x;        // (b*128 + k)*64 + d
    if (t >= KB_B * KB_N * KB_D) return;
    int b = t >> 13, rem = t & 8191, k = rem >> 6, d = rem & 63;
    size_t bb = (size_t)b * KB_N * KB_N * KB_D;
    float s = 0.f;
    for (int i = 0; i < KB_N; ++i) s += (float)X0bf[bb + ((size_t)i * KB_N + k) * KB_D + d];
    X0mean[t] = s * (1.f / KB_N);
}

__global__ void k_x7m(const float* __restrict__ A, const float* __restrict__ X0mean,
                      float* __restrict__ X7m) {
    int t = blockIdx.x * 256 + threadIdx.x;        // (b*128 + j)*64 + d
    if (t >= KB_B * KB_N * KB_D) return;
    int b = t >> 13, rem = t & 8191, j = rem >> 6, d = rem & 63;
    float s = 0.f;
    for (int k = 0; k < KB_N; ++k)
        s += A[(size_t)(b * KB_N + k) * KB_N + j] * X0mean[(b * KB_N + k) * KB_D + d];
    X7m[t] = s;
}

__global__ void k_x4diag(const float* __restrict__ A, const __bf16* __restrict__ X0bf,
                         float* __restrict__ X4diag) {
    int t = blockIdx.x * 256 + threadIdx.x;        // (b*128 + i)*64 + d
    if (t >= KB_B * KB_N * KB_D) return;
    int b = t >> 13, rem = t & 8191, i = rem >> 6, d = rem & 63;
    float s = 0.f;
    for (int k = 0; k < KB_N; ++k) {
        float a = A[(size_t)(b * KB_N + k) * KB_N + i];
        if (a != 0.f)
            s += a * (float)X0bf[((size_t)(b * KB_N + i) * KB_N + k) * KB_D + d];
    }
    X4diag[t] = s;
}

__global__ void k_cicj(const float* __restrict__ Xdiag, const float* __restrict__ Xmi,
                       const float* __restrict__ Xmj, const float* __restrict__ X7m,
                       const float* __restrict__ Wc0, const float* __restrict__ mlp1b,
                       float* __restrict__ Ci0, float* __restrict__ Cj0) {
    int t = blockIdx.x * 256 + threadIdx.x;        // (b*128 + node)*64 + n
    if (t >= KB_B * KB_N * KB_D) return;
    int n = t & 63;
    int base = t & ~63;                            // (b*128+node)*64
    float ci = mlp1b[n], cj = 0.f;
    for (int d = 0; d < 64; ++d) {
        float xd = Xdiag[base + d];
        ci += xd * Wc0[(1 * 64 + d) * 64 + n];     // X2 -> i term
        cj += xd * Wc0[(2 * 64 + d) * 64 + n];     // X3 -> j term
        ci += Xmj[base + d] * Wc0[(5 * 64 + d) * 64 + n];  // X6 -> i term
        cj += Xmi[base + d] * Wc0[(4 * 64 + d) * 64 + n];  // X5 -> j term
        cj += X7m[base + d] * Wc0[(6 * 64 + d) * 64 + n];  // X7 -> j term
    }
    Ci0[t] = ci;
    Cj0[t] = cj;
}

__global__ void k_diagfix(const float* __restrict__ Xdiag, const float* __restrict__ X4diag,
                          const float* __restrict__ Xmi, const float* __restrict__ Xmj,
                          const float* __restrict__ X7m, const float* __restrict__ Wc1,
                          const float* __restrict__ mlp1b, float* __restrict__ out) {
    int t = blockIdx.x * 256 + threadIdx.x;        // (b*128 + i)*64 + n
    if (t >= KB_B * KB_N * KB_D) return;
    int b = t >> 13, rem = t & 8191, i = rem >> 6, n = rem & 63;
    int base = t & ~63;
    float acc = mlp1b[n];
    for (int d = 0; d < 64; ++d) {
        float w123 = Wc1[(0 * 64 + d) * 64 + n] + Wc1[(1 * 64 + d) * 64 + n] +
                     Wc1[(2 * 64 + d) * 64 + n];
        acc += Xdiag[base + d] * w123;             // X1=X2=X3=Xdiag on diagonal
        acc += X4diag[base + d] * Wc1[(3 * 64 + d) * 64 + n];
        acc += Xmi[base + d]    * Wc1[(4 * 64 + d) * 64 + n];
        acc += Xmj[base + d]    * Wc1[(5 * 64 + d) * 64 + n];
        acc += X7m[base + d]    * Wc1[(6 * 64 + d) * 64 + n];
    }
    acc = acc > 0.f ? acc : 0.f;
    out[(((size_t)(b * KB_N + i)) * KB_N + i) * KB_D + n] = acc;
}

// ---------------------------------------------------------------------------
// WMMA GEMM kernels
// ---------------------------------------------------------------------------

// X0 = relu(Xbf @ lin0 + b), stored row-major bf16. M=ROWS, K=64, N=64.
__global__ void k_x0_gemm(const __bf16* __restrict__ Xbf, const __bf16* __restrict__ lin0t,
                          const float* __restrict__ lin0b, __bf16* __restrict__ X0bf) {
    int wave = (blockIdx.x << 3) + (threadIdx.x >> 5);
    int lane = threadIdx.x & 31;
    int Mbase = wave << 4;
    const __bf16* arow = Xbf + (size_t)Mbase * 64;
    v16bf a0 = load_a32(arow, 64, lane);
    v16bf a1 = load_a32(arow + 32, 64, lane);
    int half = lane >> 4, lm = lane & 15;
    for (int c = 0; c < 4; ++c) {
        int N0 = c << 4;
        v8f acc = {};
        acc = wmma_bf16(a0, load_bt32(lin0t + (size_t)N0 * 64, 64, lane), acc);
        acc = wmma_bf16(a1, load_bt32(lin0t + (size_t)N0 * 64 + 32, 64, lane), acc);
        int n = N0 + lm;
        float bias = lin0b[n];
        for (int v = 0; v < 8; ++v) {
            int row = Mbase + v + (half << 3);
            float val = acc[v] + bias;
            val = val > 0.f ? val : 0.f;
            X0bf[(size_t)row * 64 + n] = (__bf16)val;
        }
    }
}

// Y = X0 @ W4, stored TRANSPOSED per (b,i): Yt[bi][n][k]. M=ROWS, K=64, N=64.
__global__ void k_y_gemm(const __bf16* __restrict__ X0bf, const __bf16* __restrict__ W4t,
                         __bf16* __restrict__ Yt) {
    int wave = (blockIdx.x << 3) + (threadIdx.x >> 5);
    int lane = threadIdx.x & 31;
    int Mbase = wave << 4;
    const __bf16* arow = X0bf + (size_t)Mbase * 64;
    v16bf a0 = load_a32(arow, 64, lane);
    v16bf a1 = load_a32(arow + 32, 64, lane);
    int half = lane >> 4, lm = lane & 15;
    int bi = Mbase >> 7;                     // which (b,i) block of 128 rows
    int kloc = (Mbase & 127) + (half << 3);  // 8 consecutive k per lane
    for (int c = 0; c < 4; ++c) {
        int N0 = c << 4;
        v8f acc = {};
        acc = wmma_bf16(a0, load_bt32(W4t + (size_t)N0 * 64, 64, lane), acc);
        acc = wmma_bf16(a1, load_bt32(W4t + (size_t)N0 * 64 + 32, 64, lane), acc);
        int n = N0 + lm;
        v8bf pk;
        for (int v = 0; v < 8; ++v) pk[v] = (__bf16)acc[v];
        *(v8bf*)(Yt + (size_t)bi * 64 * 128 + (size_t)n * 128 + kloc) = pk;
    }
}

// out[bi, j, n] = relu( (A^T Y)[j,n] + (X W1)[j,n] + Ci0[bi,n] + Cj0[b,j,n] )
// one block per (b,i); wave w handles j in [16w, 16w+16), all 64 n.
__global__ void k_main(const __bf16* __restrict__ AbfT, const __bf16* __restrict__ Yt,
                       const __bf16* __restrict__ Xbf, const __bf16* __restrict__ W1t,
                       const float* __restrict__ Ci0, const float* __restrict__ Cj0,
                       float* __restrict__ out) {
    int bi = blockIdx.x;                  // b*128 + i
    int b = bi >> 7;
    int wave = threadIdx.x >> 5;
    int lane = threadIdx.x & 31;
    int Jbase = wave << 4;

    const __bf16* Ab = AbfT + (size_t)b * KB_N * KB_N + (size_t)Jbase * KB_N;
    v16bf a10 = load_a32(Ab, 128, lane);
    v16bf a11 = load_a32(Ab + 32, 128, lane);
    v16bf a12 = load_a32(Ab + 64, 128, lane);
    v16bf a13 = load_a32(Ab + 96, 128, lane);

    const __bf16* Xb = Xbf + (size_t)bi * KB_N * KB_D + (size_t)Jbase * KB_D;
    v16bf a20 = load_a32(Xb, 64, lane);
    v16bf a21 = load_a32(Xb + 32, 64, lane);

    const __bf16* Yb = Yt + (size_t)bi * 64 * 128;
    int half = lane >> 4, lm = lane & 15;

    for (int c = 0; c < 4; ++c) {
        int N0 = c << 4;
        const __bf16* yrow = Yb + (size_t)N0 * 128;
        v8f acc = {};
        acc = wmma_bf16(a10, load_bt32(yrow, 128, lane), acc);
        acc = wmma_bf16(a11, load_bt32(yrow + 32, 128, lane), acc);
        acc = wmma_bf16(a12, load_bt32(yrow + 64, 128, lane), acc);
        acc = wmma_bf16(a13, load_bt32(yrow + 96, 128, lane), acc);
        acc = wmma_bf16(a20, load_bt32(W1t + (size_t)N0 * 64, 64, lane), acc);
        acc = wmma_bf16(a21, load_bt32(W1t + (size_t)N0 * 64 + 32, 64, lane), acc);
        int n = N0 + lm;
        float ci = Ci0[(size_t)bi * 64 + n];
        for (int v = 0; v < 8; ++v) {
            int j = Jbase + v + (half << 3);
            float val = acc[v] + ci + Cj0[(size_t)(b * KB_N + j) * 64 + n];
            val = val > 0.f ? val : 0.f;
            out[((size_t)bi * KB_N + j) * KB_D + n] = val;
        }
    }
}

// ---------------------------------------------------------------------------
extern "C" void kernel_launch(void* const* d_in, const int* in_sizes, int n_in,
                              void* d_out, int out_size, void* d_ws, size_t ws_size,
                              hipStream_t stream) {
    const float* A     = (const float*)d_in[0];
    const float* X     = (const float*)d_in[1];
    const float* lin0w = (const float*)d_in[2];
    const float* lin0b = (const float*)d_in[3];
    const float* hw    = (const float*)d_in[4];
    const float* mlp1w = (const float*)d_in[5];
    const float* mlp1b = (const float*)d_in[6];
    float* out = (float*)d_out;

    char* ws = (char*)d_ws;
    __bf16* Xbf    = (__bf16*)(ws + OFF_XBF);
    __bf16* X0bf   = (__bf16*)(ws + OFF_X0BF);
    __bf16* Yt     = (__bf16*)(ws + OFF_YT);
    __bf16* AbfT   = (__bf16*)(ws + OFF_ABFT);
    float*  Wc0    = (float*)(ws + OFF_WC0);
    float*  Wc1    = (float*)(ws + OFF_WC1);
    __bf16* W1t    = (__bf16*)(ws + OFF_W1T);
    __bf16* W4t    = (__bf16*)(ws + OFF_W4T);
    __bf16* lin0t  = (__bf16*)(ws + OFF_LIN0T);
    float*  Xdiag  = (float*)(ws + OFF_XDIAG);
    float*  Xmi    = (float*)(ws + OFF_XMI);
    float*  Xmj    = (float*)(ws + OFF_XMJ);
    float*  X0mean = (float*)(ws + OFF_X0MEAN);
    float*  X7m    = (float*)(ws + OFF_X7M);
    float*  Ci0    = (float*)(ws + OFF_CI0);
    float*  Cj0    = (float*)(ws + OFF_CJ0);
    float*  X4diag = (float*)(ws + OFF_X4DIAG);

    const int totalX = ROWS * KB_D;          // 8,388,608
    const int bnd    = KB_B * KB_N * KB_D;   // 65,536

    k_prep_weights<<<1, 256, 0, stream>>>(hw, mlp1w, lin0w, Wc0, Wc1, W1t, W4t, lin0t);
    k_cvt_x<<<totalX / 256, 256, 0, stream>>>(X, Xbf, totalX);
    k_cvt_a<<<(KB_B * KB_N * KB_N) / 256, 256, 0, stream>>>(A, AbfT);
    k_stats1<<<bnd / 256, 256, 0, stream>>>(X, Xdiag, Xmi, Xmj);

    k_x0_gemm<<<ROWS / (8 * 16), 256, 0, stream>>>(Xbf, lin0t, lin0b, X0bf);

    k_x0mean<<<bnd / 256, 256, 0, stream>>>(X0bf, X0mean);
    k_x7m<<<bnd / 256, 256, 0, stream>>>(A, X0mean, X7m);
    k_cicj<<<bnd / 256, 256, 0, stream>>>(Xdiag, Xmi, Xmj, X7m, Wc0, mlp1b, Ci0, Cj0);

    k_y_gemm<<<ROWS / (8 * 16), 256, 0, stream>>>(X0bf, W4t, Yt);
    k_main<<<KB_B * KB_N, 256, 0, stream>>>(AbfT, Yt, Xbf, W1t, Ci0, Cj0, out);

    k_x4diag<<<bnd / 256, 256, 0, stream>>>(A, X0bf, X4diag);
    k_diagfix<<<bnd / 256, 256, 0, stream>>>(Xdiag, X4diag, Xmi, Xmj, X7m, Wc1, mlp1b, out);
}
